// LSTMDropout_39762807226938
// MI455X (gfx1250) — matrix operand order repaired and checked by reading, not compiled, not run on hardware
//
#include <hip/hip_runtime.h>
#include <hip/hip_bf16.h>

typedef __attribute__((ext_vector_type(16))) __bf16 v16bf;
typedef __attribute__((ext_vector_type(8)))  float  v8f;

#define S_LEN   1024
#define BATCH   64
#define DIN0    256
#define HID     512
#define GATES   2048      // 4*H
#define INV_KEEP 1.25f    // 1/(1-0.2)

struct alignas(16) Q4 { unsigned int u[4]; };
struct alignas(32) Pack32 { Q4 a, b; };

__device__ __forceinline__ unsigned short f2bf(float f) {
  unsigned int u = __builtin_bit_cast(unsigned int, f);
  u += 0x7fffu + ((u >> 16) & 1u);
  return (unsigned short)(u >> 16);
}

__device__ __forceinline__ float sigmoidf_(float x) {
  return 1.0f / (1.0f + __expf(-x));
}

// A operand: 16x32 bf16 tile, per-lane layout (K 0-7/16-23 vs 8-15/24-31 halves)
__device__ __forceinline__ v16bf ldA(const unsigned short* base, int half) {
  Pack32 p;
  p.a = *reinterpret_cast<const Q4*>(base + 8 * half);
  p.b = *reinterpret_cast<const Q4*>(base + 16 + 8 * half);
  return __builtin_bit_cast(v16bf, p);
}

// B operand: K 0-7/16-23 vs 8-15/24-31 halves per lane (base already includes 8*half)
__device__ __forceinline__ v16bf ldB(const unsigned short* base) {
  Pack32 p;
  p.a = *reinterpret_cast<const Q4*>(base);
  p.b = *reinterpret_cast<const Q4*>(base + 16);
  return __builtin_bit_cast(v16bf, p);
}

__device__ __forceinline__ v8f wmma_bf16(v16bf a, v16bf b, v8f c) {
  return __builtin_amdgcn_wmma_f32_16x16x32_bf16(false, a, false, b, (short)0, c,
                                                 false, false);
}

// ---------------- prep kernels ----------------

__global__ void k_f32_to_bf16(const float* __restrict__ src,
                              unsigned short* __restrict__ dst, int n) {
  for (int i = blockIdx.x * blockDim.x + threadIdx.x; i < n;
       i += gridDim.x * blockDim.x)
    dst[i] = f2bf(src[i]);
}

__global__ void k_pack_wcat(const float* __restrict__ Wih,
                            const float* __restrict__ Whh,
                            unsigned short* __restrict__ Wcat,
                            int Din, int n) {
  const int K = Din + HID;
  for (int i = blockIdx.x * blockDim.x + threadIdx.x; i < n;
       i += gridDim.x * blockDim.x) {
    int g = i / K, k = i - g * K;
    float v = (k < Din) ? Wih[(size_t)g * Din + k]
                        : Whh[(size_t)g * HID + (k - Din)];
    Wcat[i] = f2bf(v);
  }
}

__global__ void k_bias(const float* __restrict__ bi, const float* __restrict__ bh,
                       float* __restrict__ o, int n) {
  for (int i = blockIdx.x * blockDim.x + threadIdx.x; i < n;
       i += gridDim.x * blockDim.x)
    o[i] = bi[i] + bh[i];
}

__global__ void k_init(const float* __restrict__ h0, const float* __restrict__ c0,
                       const unsigned char* __restrict__ mh,
                       unsigned short* __restrict__ hdrop,
                       float* __restrict__ cst, int n) {
  for (int i = blockIdx.x * blockDim.x + threadIdx.x; i < n;
       i += gridDim.x * blockDim.x) {
    float kh = mh[i] ? 0.0f : INV_KEEP;
    hdrop[i] = f2bf(h0[i] * kh);
    cst[i]   = c0[i];
  }
}

__global__ void k_finalize(const float* __restrict__ last_h,
                           const float* __restrict__ c_state,
                           float* __restrict__ h_dst,
                           float* __restrict__ c_dst, int n) {
  for (int i = blockIdx.x * blockDim.x + threadIdx.x; i < n;
       i += gridDim.x * blockDim.x) {
    h_dst[i] = last_h[i];
    c_dst[i] = c_state[i];
  }
}

// ---------------- one LSTM timestep ----------------
// gates = [x_t | h_drop] * Wcat^T + (b_ih+b_hh); software-pipelined WMMA K-loop,
// then in-register i/f/g/o + c/h update + dropout-scaled bf16 h for next step.
template <int DIN>
__global__ __launch_bounds__(256)
void k_lstm_step(const unsigned short* __restrict__ xsrc,
                 const unsigned short* __restrict__ Wcat,
                 const float* __restrict__ bias,
                 const unsigned short* __restrict__ hdrop_in,
                 unsigned short* __restrict__ hdrop_out,
                 float* __restrict__ c_state,
                 const unsigned char* __restrict__ mask_h,
                 const unsigned char* __restrict__ mask_c,
                 float* __restrict__ out_f32,
                 unsigned short* __restrict__ out_bf16) {
  constexpr int K   = DIN + HID;
  constexpr int NBX = DIN >> 5;   // k-blocks in x region
  constexpr int NKB = K >> 5;     // total k-blocks
  const int lane = threadIdx.x & 31;
  const int wave = blockIdx.x * (blockDim.x >> 5) + (threadIdx.x >> 5); // 0..127
  const int m0   = (wave & 3) * 16;   // batch row tile
  const int n0   = (wave >> 2) * 16;  // hidden col tile
  const int half = lane >> 4;
  const int l16  = lane & 15;

  v8f acc[4];
#pragma unroll
  for (int g = 0; g < 4; ++g) {
    float bv = bias[g * HID + n0 + l16];
#pragma unroll
    for (int r = 0; r < 8; ++r) acc[g][r] = bv;
  }

  const int am = m0 + l16;
  const unsigned short* ax = xsrc + (size_t)am * DIN;       // x region base
  const unsigned short* ah = hdrop_in + (size_t)am * HID;   // h region base
  const unsigned short* bb0 = Wcat + (size_t)(0 * HID + n0 + l16) * K + 8 * half;
  const unsigned short* bb1 = Wcat + (size_t)(1 * HID + n0 + l16) * K + 8 * half;
  const unsigned short* bb2 = Wcat + (size_t)(2 * HID + n0 + l16) * K + 8 * half;
  const unsigned short* bb3 = Wcat + (size_t)(3 * HID + n0 + l16) * K + 8 * half;

  // ---- software-pipelined K loop (fully unrolled, immediate offsets) ----
  v16bf aC = ldA(ax, half);
  v16bf b0 = ldB(bb0), b1 = ldB(bb1), b2 = ldB(bb2), b3 = ldB(bb3);
#pragma unroll
  for (int kb = 1; kb < NKB; ++kb) {
    const unsigned short* an =
        (kb < NBX) ? (ax + kb * 32) : (ah + (kb - NBX) * 32);
    v16bf aN  = ldA(an, half);
    v16bf n0b = ldB(bb0 + kb * 32);
    v16bf n1b = ldB(bb1 + kb * 32);
    v16bf n2b = ldB(bb2 + kb * 32);
    v16bf n3b = ldB(bb3 + kb * 32);
    acc[0] = wmma_bf16(aC, b0, acc[0]);
    acc[1] = wmma_bf16(aC, b1, acc[1]);
    acc[2] = wmma_bf16(aC, b2, acc[2]);
    acc[3] = wmma_bf16(aC, b3, acc[3]);
    aC = aN; b0 = n0b; b1 = n1b; b2 = n2b; b3 = n3b;
  }
  acc[0] = wmma_bf16(aC, b0, acc[0]);
  acc[1] = wmma_bf16(aC, b1, acc[1]);
  acc[2] = wmma_bf16(aC, b2, acc[2]);
  acc[3] = wmma_bf16(aC, b3, acc[3]);

  // ---- elementwise LSTM cell update; exclusive (b,h) ownership per lane ----
  const int hidx = n0 + l16;
#pragma unroll
  for (int r = 0; r < 8; ++r) {
    const int brow = m0 + r + 8 * half;   // C/D layout: M = r + 8*(lane>=16)
    const int idx  = brow * HID + hidx;
    float iv = sigmoidf_(acc[0][r]);
    float fv = sigmoidf_(acc[1][r]);
    float gv = tanhf(acc[2][r]);
    float ov = sigmoidf_(acc[3][r]);
    float kc = mask_c[idx] ? 0.0f : INV_KEEP;
    float cn = fv * (c_state[idx] * kc) + iv * gv;
    c_state[idx] = cn;
    float hv = ov * tanhf(cn);
    if (out_f32)  out_f32[idx]  = hv;
    if (out_bf16) out_bf16[idx] = f2bf(hv);
    float kh = mask_h[idx] ? 0.0f : INV_KEEP;
    hdrop_out[idx] = f2bf(hv * kh);
  }
}

// ---------------- host ----------------

extern "C" void kernel_launch(void* const* d_in, const int* in_sizes, int n_in,
                              void* d_out, int out_size, void* d_ws, size_t ws_size,
                              hipStream_t stream) {
  (void)in_sizes; (void)n_in; (void)out_size; (void)ws_size;

  const float* x    = (const float*)d_in[0];
  const float* Wih0 = (const float*)d_in[1];
  const float* Whh0 = (const float*)d_in[2];
  const float* bih0 = (const float*)d_in[3];
  const float* bhh0 = (const float*)d_in[4];
  const float* h00  = (const float*)d_in[5];
  const float* c00  = (const float*)d_in[6];
  const unsigned char* mh0 = (const unsigned char*)d_in[7];
  const unsigned char* mc0 = (const unsigned char*)d_in[8];
  const float* Wih1 = (const float*)d_in[9];
  const float* Whh1 = (const float*)d_in[10];
  const float* bih1 = (const float*)d_in[11];
  const float* bhh1 = (const float*)d_in[12];
  const float* h01  = (const float*)d_in[13];
  const float* c01  = (const float*)d_in[14];
  const unsigned char* mh1 = (const unsigned char*)d_in[15];
  const unsigned char* mc1 = (const unsigned char*)d_in[16];

  char* w = (char*)d_ws;
  size_t off = 0;
  auto take = [&](size_t bytes) -> void* {
    void* p = w + off;
    off = (off + bytes + 255) & ~(size_t)255;
    return p;
  };

  const int BH = BATCH * HID;
  unsigned short* x_bf  = (unsigned short*)take((size_t)S_LEN * BATCH * DIN0 * 2);
  unsigned short* o0_bf = (unsigned short*)take((size_t)S_LEN * BH * 2);
  unsigned short* Wcat0 = (unsigned short*)take((size_t)GATES * (DIN0 + HID) * 2);
  unsigned short* Wcat1 = (unsigned short*)take((size_t)GATES * (HID + HID) * 2);
  float* bias0 = (float*)take((size_t)GATES * 4);
  float* bias1 = (float*)take((size_t)GATES * 4);
  unsigned short* hd0 = (unsigned short*)take((size_t)2 * BH * 2);
  unsigned short* hd1 = (unsigned short*)take((size_t)2 * BH * 2);
  float* cst0 = (float*)take((size_t)BH * 4);
  float* cst1 = (float*)take((size_t)BH * 4);

  float* out1  = (float*)d_out;
  float* h_dst = out1 + (size_t)S_LEN * BH;
  float* c_dst = h_dst + BH;

  // prep
  k_f32_to_bf16<<<4096, 256, 0, stream>>>(x, x_bf, S_LEN * BATCH * DIN0);
  k_pack_wcat<<<2048, 256, 0, stream>>>(Wih0, Whh0, Wcat0, DIN0, GATES * (DIN0 + HID));
  k_pack_wcat<<<2048, 256, 0, stream>>>(Wih1, Whh1, Wcat1, HID, GATES * (HID + HID));
  k_bias<<<8, 256, 0, stream>>>(bih0, bhh0, bias0, GATES);
  k_bias<<<8, 256, 0, stream>>>(bih1, bhh1, bias1, GATES);
  k_init<<<128, 256, 0, stream>>>(h00, c00, mh0, hd0, cst0, BH);
  k_init<<<128, 256, 0, stream>>>(h01, c01, mh1, hd1, cst1, BH);

  // layer 0: writes bf16 hidden sequence for layer 1
  for (int t = 0; t < S_LEN; ++t) {
    k_lstm_step<DIN0><<<16, 256, 0, stream>>>(
        x_bf + (size_t)t * BATCH * DIN0, Wcat0, bias0,
        hd0 + (t & 1) * BH, hd0 + ((t + 1) & 1) * BH, cst0, mh0, mc0,
        (float*)nullptr, o0_bf + (size_t)t * BH);
  }
  // layer 1: writes fp32 outputs straight into d_out
  for (int t = 0; t < S_LEN; ++t) {
    k_lstm_step<HID><<<16, 256, 0, stream>>>(
        o0_bf + (size_t)t * BH, Wcat1, bias1,
        hd1 + (t & 1) * BH, hd1 + ((t + 1) & 1) * BH, cst1, mh1, mc1,
        out1 + (size_t)t * BH, (unsigned short*)nullptr);
  }
  k_finalize<<<128, 256, 0, stream>>>(out1 + (size_t)(S_LEN - 1) * BH, cst1,
                                      h_dst, c_dst, BH);
}
